// ADIAModel_43198781063569
// MI455X (gfx1250) — matrix operand order repaired and checked by reading, not compiled, not run on hardware
//
#include <hip/hip_runtime.h>
#include <hip/hip_bf16.h>

#define B_   16
#define E_   240
#define C_   8
#define N_   256
#define D_   64
#define NB_  5
#define NL_  2
#define H_   4
#define DH_  16
#define M_   14
#define BE_  (B_*E_)   /* 3840 */

typedef __attribute__((ext_vector_type(16))) __bf16 v16bf;
typedef __attribute__((ext_vector_type(8)))  float  v8f;

union BFrag {            // B operand assembled from two aligned 16B LDS loads
  v16bf v;
  uint4 q[2];
};

__device__ __forceinline__ float gelu_exact(float x) {
  return 0.5f * x * (1.0f + erff(x * 0.70710678118654752f));
}

// ---------------------------------------------------------------------------
// Kernel 1: stem + 5x (conv1d k=3 via bf16 WMMA, GroupNorm(8), GELU, residual)
// One block per edge sequence m in [0, 3840). 256 threads = 8 wave32 waves.
// LDS: fp32 master x[64][256] + bf16 transposed shadow xbT[258][64] (conv halo).
// ---------------------------------------------------------------------------
__global__ __launch_bounds__(256)
void conv_tower_kernel(const float* __restrict__ edge_data,
                       const float* __restrict__ stem_w,
                       const float* __restrict__ stem_b,
                       const float* __restrict__ conv_w,
                       const float* __restrict__ gn_g,
                       const float* __restrict__ gn_b,
                       float* __restrict__ conv_emb) {
  __shared__ float xf[D_ * N_];                          // 64 KB
  __shared__ __align__(32) __bf16 xbT[(N_ + 2) * D_];    // 33 KB, [col][chan]
  __shared__ float gsum[8], gssq[8];
  __shared__ float ce[D_];

  const int m      = blockIdx.x;
  const int tid    = threadIdx.x;
  const int wave   = tid >> 5;
  const int lane   = tid & 31;
  const int hi     = lane >> 4;     // half-wave select per WMMA layouts
  const int l15    = lane & 15;
  const int dt     = wave & 3;      // output-channel tile (rows 16*dt..16*dt+15)
  const int ntBase = (wave >> 2) * 8;

  // ---- stem: x[d][n] = sum_c ed[c][n]*stem_w[c][d] + stem_b[d] ----
  const float* ed = edge_data + (size_t)m * (C_ * N_);
  for (int e = tid; e < D_ * N_; e += 256) {
    int d = e >> 8, n = e & 255;
    float v = stem_b[d];
#pragma unroll
    for (int c = 0; c < C_; ++c) v += ed[c * N_ + n] * stem_w[c * D_ + d];
    xf[d * N_ + n] = v;
    xbT[(n + 1) * D_ + d] = (__bf16)v;
  }
  if (tid < D_) {                                        // conv halo columns
    xbT[0 * D_ + tid] = (__bf16)0.f;
    xbT[(N_ + 1) * D_ + tid] = (__bf16)0.f;
  }
  __syncthreads();

  for (int blk = 0; blk < NB_; ++blk) {
    // ---- A fragments (16x32 bf16): lane l15 = row M, K = j + 16*hi (contiguous, same as B) ----
    v16bf afr[3][2];
    const float* wb = conv_w + (size_t)blk * (D_ * D_ * 3);
    const int dout = dt * 16 + l15;
#pragma unroll
    for (int k = 0; k < 3; ++k)
#pragma unroll
      for (int cb = 0; cb < 2; ++cb)
#pragma unroll
        for (int j = 0; j < 16; ++j) {
          int K   = j + 16 * hi;
          int din = cb * 32 + K;
          afr[k][cb][j] = (__bf16)wb[dout * (D_ * 3) + din * 3 + k];
        }

    if (tid < 16) { gsum[tid & 7] = 0.f; gssq[tid & 7] = 0.f; }
    __syncthreads();

    // ---- 8 output tiles per wave: 6 WMMAs each (3 taps x 2 K-blocks) ----
    v8f acc[8];
#pragma unroll
    for (int t = 0; t < 8; ++t) {
      const int nt = ntBase + t;
      v8f c8 = {0.f, 0.f, 0.f, 0.f, 0.f, 0.f, 0.f, 0.f};
#pragma unroll
      for (int k = 0; k < 3; ++k) {
#pragma unroll
        for (int cb = 0; cb < 2; ++cb) {
          BFrag bfr;  // B 32x16 bf16: elem j -> (K = j + 16*hi, N = l15)
          const uint4* p = (const uint4*)
              &xbT[(nt * 16 + l15 + k) * D_ + cb * 32 + 16 * hi];
          bfr.q[0] = p[0];
          bfr.q[1] = p[1];
          c8 = __builtin_amdgcn_wmma_f32_16x16x32_bf16(
              false, afr[k][cb], false, bfr.v, (short)0, c8, false, false);
        }
      }
      acc[t] = c8;
    }

    // ---- GroupNorm stats: this thread's 64 outputs all lie in one group ----
    const int g = 2 * dt + hi;
    float s = 0.f, q = 0.f;
#pragma unroll
    for (int t = 0; t < 8; ++t)
#pragma unroll
      for (int r = 0; r < 8; ++r) { float v = acc[t][r]; s += v; q += v * v; }
    atomicAdd(&gsum[g], s);
    atomicAdd(&gssq[g], q);
    __syncthreads();

    const float mu  = gsum[g] * (1.f / 2048.f);
    const float var = gssq[g] * (1.f / 2048.f) - mu * mu;
    const float inv = rsqrtf(var + 1e-5f);
#pragma unroll
    for (int t = 0; t < 8; ++t) {
      const int n = (ntBase + t) * 16 + l15;
#pragma unroll
      for (int r = 0; r < 8; ++r) {
        const int c  = dt * 16 + r + 8 * hi;     // C-layout: M = r + 8*hi
        float hn = (acc[t][r] - mu) * inv * gn_g[blk * D_ + c] + gn_b[blk * D_ + c];
        float xv = xf[c * N_ + n] + gelu_exact(hn);
        xf[c * N_ + n] = xv;
        xbT[(n + 1) * D_ + c] = (__bf16)xv;
      }
    }
    __syncthreads();
  }

  // ---- conv_emb = mean over N ----
  if (tid < D_) ce[tid] = 0.f;
  __syncthreads();
  {
    int d = tid >> 2, part = tid & 3;
    float s = 0.f;
    for (int n = part * 64; n < part * 64 + 64; ++n) s += xf[d * N_ + n];
    atomicAdd(&ce[d], s);
  }
  __syncthreads();
  if (tid < D_) conv_emb[(size_t)m * D_ + tid] = ce[tid] * (1.f / 256.f);
}

// ---------------------------------------------------------------------------
// Kernel 2: edge embedding: gelu(ln(concat(conv_emb, type_emb) @ em_w + em_b))
// ---------------------------------------------------------------------------
__global__ __launch_bounds__(64)
void edge_embed_kernel(const float* __restrict__ conv_emb,
                       const int* __restrict__ edge_types,
                       const float* __restrict__ type_tab,
                       const float* __restrict__ em_w, const float* __restrict__ em_b,
                       const float* __restrict__ em_g, const float* __restrict__ em_be,
                       float* __restrict__ emb) {
  __shared__ float cerow[64], terow[64], red[2];
  const int m = blockIdx.x, d = threadIdx.x;
  const int ty = edge_types[m];
  cerow[d] = conv_emb[(size_t)m * 64 + d];
  terow[d] = type_tab[ty * 64 + d];
  if (d == 0) { red[0] = 0.f; red[1] = 0.f; }
  __syncthreads();
  float h = em_b[d];
  for (int j = 0; j < 64; ++j)
    h += cerow[j] * em_w[j * 64 + d] + terow[j] * em_w[(64 + j) * 64 + d];
  atomicAdd(&red[0], h); atomicAdd(&red[1], h * h);
  __syncthreads();
  const float mu = red[0] * (1.f / 64.f);
  const float var = red[1] * (1.f / 64.f) - mu * mu;
  float v = (h - mu) * rsqrtf(var + 1e-5f) * em_g[d] + em_be[d];
  emb[(size_t)m * 64 + d] = gelu_exact(v);
}

// ---------------------------------------------------------------------------
// Kernel 3a: qkv = emb @ qkv_w[l] + qkv_b[l]
// ---------------------------------------------------------------------------
__global__ __launch_bounds__(192)
void qkv_kernel(const float* __restrict__ emb, const float* __restrict__ qkv_w,
                const float* __restrict__ qkv_b, int l, float* __restrict__ qkv) {
  __shared__ float er[64];
  const int m = blockIdx.x, j = threadIdx.x;
  if (j < 64) er[j] = emb[(size_t)m * 64 + j];
  __syncthreads();
  const float* W = qkv_w + (size_t)l * 64 * 192;
  float v = qkv_b[l * 192 + j];
  for (int c = 0; c < 64; ++c) v += er[c] * W[c * 192 + j];
  qkv[(size_t)m * 192 + j] = v;
}

// ---------------------------------------------------------------------------
// Kernel 3b: attention core, one block per (batch, head); wave-per-row softmax
// ---------------------------------------------------------------------------
__global__ __launch_bounds__(256)
void attn_kernel(const float* __restrict__ qkv, const unsigned char* __restrict__ mask,
                 float* __restrict__ obuf) {
  __shared__ float qh[E_ * 16], kh[E_ * 16], vh[E_ * 16], mb[E_];
  const int b = blockIdx.x >> 2, h = blockIdx.x & 3;
  const int tid = threadIdx.x, wave = tid >> 5, lane = tid & 31;
  for (int idx = tid; idx < E_ * 16; idx += 256) {
    int e = idx >> 4, d = idx & 15;
    size_t base = ((size_t)b * E_ + e) * 192 + h * 16 + d;
    qh[idx] = qkv[base];
    kh[idx] = qkv[base + 64];
    vh[idx] = qkv[base + 128];
  }
  if (tid < E_) mb[tid] = mask[b * E_ + tid] ? 0.f : -1e9f;
  __syncthreads();

  for (int e = wave; e < E_; e += 8) {
    float sc[8];
    float mx = -1e30f;
#pragma unroll
    for (int i = 0; i < 8; ++i) {
      int k = lane + 32 * i;
      float s = -1e30f;
      if (k < E_) {
        s = 0.f;
#pragma unroll
        for (int d = 0; d < 16; ++d) s += qh[e * 16 + d] * kh[k * 16 + d];
        s = s * 0.25f + mb[k];   // 1/sqrt(16)
      }
      sc[i] = s;
      mx = fmaxf(mx, s);
    }
#pragma unroll
    for (int off = 16; off > 0; off >>= 1) mx = fmaxf(mx, __shfl_xor(mx, off, 32));
    float se = 0.f;
#pragma unroll
    for (int i = 0; i < 8; ++i) { float p = expf(sc[i] - mx); sc[i] = p; se += p; }
#pragma unroll
    for (int off = 16; off > 0; off >>= 1) se += __shfl_xor(se, off, 32);
    float oa[16];
#pragma unroll
    for (int d = 0; d < 16; ++d) oa[d] = 0.f;
#pragma unroll
    for (int i = 0; i < 8; ++i) {
      int k = lane + 32 * i;
      if (k < E_) {
        float p = sc[i];
#pragma unroll
        for (int d = 0; d < 16; ++d) oa[d] += p * vh[k * 16 + d];
      }
    }
    const float inv = 1.f / se;
#pragma unroll
    for (int d = 0; d < 16; ++d) {
      float r = oa[d];
#pragma unroll
      for (int off = 16; off > 0; off >>= 1) r += __shfl_xor(r, off, 32);
      if (lane == 0) obuf[((size_t)b * E_ + e) * 64 + h * 16 + d] = r * inv;
    }
  }
}

// ---------------------------------------------------------------------------
// Kernel 3c: o-projection + residual + LayerNorm1 (writes emb in place)
// ---------------------------------------------------------------------------
__global__ __launch_bounds__(64)
void proj_ln_kernel(const float* __restrict__ obuf, const float* __restrict__ out_w,
                    const float* __restrict__ out_b, const float* __restrict__ n1_g,
                    const float* __restrict__ n1_b, int l, float* __restrict__ emb) {
  __shared__ float orow[64], red[2];
  const int m = blockIdx.x, d = threadIdx.x;
  orow[d] = obuf[(size_t)m * 64 + d];
  if (d == 0) { red[0] = 0.f; red[1] = 0.f; }
  __syncthreads();
  const float* W = out_w + (size_t)l * 64 * 64;
  float v = out_b[l * 64 + d];
  for (int c = 0; c < 64; ++c) v += orow[c] * W[c * 64 + d];
  v += emb[(size_t)m * 64 + d];
  atomicAdd(&red[0], v); atomicAdd(&red[1], v * v);
  __syncthreads();
  const float mu = red[0] * (1.f / 64.f);
  const float var = red[1] * (1.f / 64.f) - mu * mu;
  emb[(size_t)m * 64 + d] = (v - mu) * rsqrtf(var + 1e-5f) * n1_g[l * 64 + d] + n1_b[l * 64 + d];
}

// ---------------------------------------------------------------------------
// Kernel 3d: ff1 = gelu(emb @ ff1_w[l] + ff1_b[l])
// ---------------------------------------------------------------------------
__global__ __launch_bounds__(256)
void ff1_kernel(const float* __restrict__ emb, const float* __restrict__ ff1_w,
                const float* __restrict__ ff1_b, int l, float* __restrict__ f1) {
  __shared__ float er[64];
  const int m = blockIdx.x, j = threadIdx.x;
  if (j < 64) er[j] = emb[(size_t)m * 64 + j];
  __syncthreads();
  const float* W = ff1_w + (size_t)l * 64 * 256;
  float v = ff1_b[l * 256 + j];
  for (int c = 0; c < 64; ++c) v += er[c] * W[c * 256 + j];
  f1[(size_t)m * 256 + j] = gelu_exact(v);
}

// ---------------------------------------------------------------------------
// Kernel 3e: ff2 + residual + LayerNorm2 (writes emb in place)
// ---------------------------------------------------------------------------
__global__ __launch_bounds__(64)
void ff2_ln_kernel(const float* __restrict__ f1, const float* __restrict__ ff2_w,
                   const float* __restrict__ ff2_b, const float* __restrict__ n2_g,
                   const float* __restrict__ n2_b, int l, float* __restrict__ emb) {
  __shared__ float fr[256], red[2];
  const int m = blockIdx.x, d = threadIdx.x;
  for (int t = d; t < 256; t += 64) fr[t] = f1[(size_t)m * 256 + t];
  if (d == 0) { red[0] = 0.f; red[1] = 0.f; }
  __syncthreads();
  const float* W = ff2_w + (size_t)l * 256 * 64;
  float v = ff2_b[l * 64 + d];
  for (int j = 0; j < 256; ++j) v += fr[j] * W[j * 64 + d];
  v += emb[(size_t)m * 64 + d];
  atomicAdd(&red[0], v); atomicAdd(&red[1], v * v);
  __syncthreads();
  const float mu = red[0] * (1.f / 64.f);
  const float var = red[1] * (1.f / 64.f) - mu * mu;
  emb[(size_t)m * 64 + d] = (v - mu) * rsqrtf(var + 1e-5f) * n2_g[l * 64 + d] + n2_b[l * 64 + d];
}

// ---------------------------------------------------------------------------
// Kernel 4: edge head -> d_out[0 .. 7680)
// ---------------------------------------------------------------------------
__global__ __launch_bounds__(256)
void edge_head_kernel(const float* __restrict__ emb, const float* __restrict__ ehw,
                      const float* __restrict__ ehb, float* __restrict__ out) {
  const int i = blockIdx.x * 256 + threadIdx.x;
  if (i >= BE_ * 2) return;
  const int m = i >> 1, o = i & 1;
  float v = ehb[o];
  for (int d = 0; d < 64; ++d) v += emb[(size_t)m * 64 + d] * ehw[d * 2 + o];
  out[i] = v;
}

// ---------------------------------------------------------------------------
// Kernel 5: node gather + MLP + head -> d_out[7680 .. 9472)
// ---------------------------------------------------------------------------
__global__ __launch_bounds__(64)
void node_kernel(const float* __restrict__ emb, const int* __restrict__ nidx,
                 const float* __restrict__ nm_w, const float* __restrict__ nm_b,
                 const float* __restrict__ nm_g, const float* __restrict__ nm_be,
                 const float* __restrict__ nh_w, const float* __restrict__ nh_b,
                 float* __restrict__ out) {
  __shared__ float g[256], ne[64], red[2];
  const int bv = blockIdx.x;            // b*14 + v
  const int b = bv / M_;
  const int d = threadIdx.x;
  for (int t = d; t < 256; t += 64) {
    int jj = t >> 6, c = t & 63;
    int ei = nidx[bv * 4 + jj];
    g[t] = emb[((size_t)b * E_ + ei) * 64 + c];
  }
  if (d == 0) { red[0] = 0.f; red[1] = 0.f; }
  __syncthreads();
  float h = nm_b[d];
  for (int t = 0; t < 256; ++t) h += g[t] * nm_w[t * 64 + d];
  atomicAdd(&red[0], h); atomicAdd(&red[1], h * h);
  __syncthreads();
  const float mu = red[0] * (1.f / 64.f);
  const float var = red[1] * (1.f / 64.f) - mu * mu;
  ne[d] = gelu_exact((h - mu) * rsqrtf(var + 1e-5f) * nm_g[d] + nm_be[d]);
  __syncthreads();
  if (d < 8) {
    float v = nh_b[d];
    for (int c = 0; c < 64; ++c) v += ne[c] * nh_w[c * 8 + d];
    out[BE_ * 2 + bv * 8 + d] = v;
  }
}

// ---------------------------------------------------------------------------
extern "C" void kernel_launch(void* const* d_in, const int* in_sizes, int n_in,
                              void* d_out, int out_size, void* d_ws, size_t ws_size,
                              hipStream_t stream) {
  (void)in_sizes; (void)n_in; (void)out_size; (void)ws_size;

  const float*        edge_data   = (const float*)d_in[0];
  const int*          edge_types  = (const int*)d_in[1];
  const unsigned char* edge_mask  = (const unsigned char*)d_in[2];
  const int*          node_eidx   = (const int*)d_in[3];
  const float* stem_w  = (const float*)d_in[4];
  const float* stem_b  = (const float*)d_in[5];
  const float* conv_w  = (const float*)d_in[6];
  const float* gn_g    = (const float*)d_in[7];
  const float* gn_b    = (const float*)d_in[8];
  const float* type_tab= (const float*)d_in[9];
  const float* em_w    = (const float*)d_in[10];
  const float* em_b    = (const float*)d_in[11];
  const float* em_g    = (const float*)d_in[12];
  const float* em_be   = (const float*)d_in[13];
  const float* qkv_w   = (const float*)d_in[14];
  const float* qkv_b   = (const float*)d_in[15];
  const float* out_w   = (const float*)d_in[16];
  const float* out_b   = (const float*)d_in[17];
  const float* n1_g    = (const float*)d_in[18];
  const float* n1_b    = (const float*)d_in[19];
  const float* ff1_w   = (const float*)d_in[20];
  const float* ff1_b   = (const float*)d_in[21];
  const float* ff2_w   = (const float*)d_in[22];
  const float* ff2_b   = (const float*)d_in[23];
  const float* n2_g    = (const float*)d_in[24];
  const float* n2_b    = (const float*)d_in[25];
  const float* ehw     = (const float*)d_in[26];
  const float* ehb     = (const float*)d_in[27];
  const float* nm_w    = (const float*)d_in[28];
  const float* nm_b    = (const float*)d_in[29];
  const float* nm_g    = (const float*)d_in[30];
  const float* nm_be   = (const float*)d_in[31];
  const float* nh_w    = (const float*)d_in[32];
  const float* nh_b    = (const float*)d_in[33];

  float* ws = (float*)d_ws;
  float* ws_conv_emb = ws;                      // 3840*64
  float* ws_emb      = ws + (size_t)BE_ * 64;   // 3840*64
  float* ws_qkv      = ws + (size_t)BE_ * 128;  // 3840*192
  float* ws_obuf     = ws + (size_t)BE_ * 320;  // 3840*64
  float* ws_f1       = ws + (size_t)BE_ * 384;  // 3840*256  (total 640*BE_ floats)
  float* out         = (float*)d_out;

  conv_tower_kernel<<<BE_, 256, 0, stream>>>(edge_data, stem_w, stem_b, conv_w,
                                             gn_g, gn_b, ws_conv_emb);
  edge_embed_kernel<<<BE_, 64, 0, stream>>>(ws_conv_emb, edge_types, type_tab,
                                            em_w, em_b, em_g, em_be, ws_emb);
  for (int l = 0; l < NL_; ++l) {
    qkv_kernel   <<<BE_, 192, 0, stream>>>(ws_emb, qkv_w, qkv_b, l, ws_qkv);
    attn_kernel  <<<B_ * H_, 256, 0, stream>>>(ws_qkv, edge_mask, ws_obuf);
    proj_ln_kernel<<<BE_, 64, 0, stream>>>(ws_obuf, out_w, out_b, n1_g, n1_b, l, ws_emb);
    ff1_kernel   <<<BE_, 256, 0, stream>>>(ws_emb, ff1_w, ff1_b, l, ws_f1);
    ff2_ln_kernel<<<BE_, 64, 0, stream>>>(ws_f1, ff2_w, ff2_b, n2_g, n2_b, l, ws_emb);
  }
  edge_head_kernel<<<(BE_ * 2 + 255) / 256, 256, 0, stream>>>(ws_emb, ehw, ehb, out);
  node_kernel<<<B_ * M_, 64, 0, stream>>>(ws_emb, node_eidx, nm_w, nm_b, nm_g, nm_be,
                                          nh_w, nh_b, out);
}
